// MK_MMDLoss_18734647345271
// MI455X (gfx1250) — matrix operand order repaired and checked by reading, not compiled, not run on hardware
//
#include <hip/hip_runtime.h>
#include <hip/hip_bf16.h>

typedef __bf16 bf16;
typedef __attribute__((ext_vector_type(16))) __bf16 v16bf;
typedef __attribute__((ext_vector_type(8)))  __bf16 v8bf;
typedef __attribute__((ext_vector_type(8)))  float  v8f;

union V16U { v16bf v; v8bf h[2]; };

#define NROWS 4096   // B*H*W
#define CDIM  256
#define KD    1024   // 4*C after hi/lo expansion

#define BM 256       // block tile rows
#define BN 128       // block tile cols
#define GX (NROWS / BM)   // 16
#define GY (NROWS / BN)   // 32
#define NBLK (GX * GY)    // 512 blocks per matrix

// ---------------------------------------------------------------------------
// Pack x (B,C,H,W) fp32 -> row-major [N][KD] bf16 in A-form and B-form.
// Row n = b*1024 + p (p = h*32+w). k' = 4c+t:
//   A-form: [hi,hi,lo,lo]   B-form: [hi,lo,hi,lo]
// so sum_k' A[i][k']*B[j][k'] = (hi_a+lo_a)·(hi_b+lo_b) ~= fp32 dot.
// ---------------------------------------------------------------------------
__global__ void pack_kernel(const float* __restrict__ src,
                            bf16* __restrict__ Aout, bf16* __restrict__ Bout) {
    int tid = blockIdx.x * blockDim.x + threadIdx.x;   // 0 .. 4096*256-1
    int c = tid & (CDIM - 1);
    int n = tid >> 8;
    int b = n >> 10, p = n & 1023;
    float f = src[(((size_t)b * CDIM + c) << 10) + p];
    bf16 hi = (bf16)f;
    bf16 lo = (bf16)(f - (float)hi);
    size_t base = ((size_t)n << 10) + 4 * c;
    Aout[base + 0] = hi; Aout[base + 1] = hi; Aout[base + 2] = lo; Aout[base + 3] = lo;
    Bout[base + 0] = hi; Bout[base + 1] = lo; Bout[base + 2] = hi; Bout[base + 3] = lo;
}

// Row squared norms in fp32 (exact path for the a2+b2 part of the distance).
__global__ void norm_kernel(const float* __restrict__ src, float* __restrict__ nrm) {
    int n = blockIdx.x * blockDim.x + threadIdx.x;     // 0..4095
    int b = n >> 10, p = n & 1023;
    const float* col = src + (((size_t)b * CDIM) << 10) + p;
    float s = 0.f;
#pragma unroll 4
    for (int c = 0; c < CDIM; ++c) {
        float v = col[(size_t)c << 10];
        s = fmaf(v, v, s);
    }
    nrm[n] = s;
}

// ---------------------------------------------------------------------------
// 128 threads = 4 waves (2x2); each wave computes a 128x64 tile (8x4 WMMA
// fragments) of the Gram matrix, then folds the Gaussian epilogue.
// blockIdx.z: 0=xx, 1=yy, 2=xy. xx/yy use triangular weighting: blocks
// entirely above the diagonal are skipped; remaining elements get weight
// (i>j ? 2 : i==j ? 1 : 0), which equals the full symmetric sum.
// ---------------------------------------------------------------------------
__global__ __launch_bounds__(128)
void mmd_tile_kernel(const bf16* __restrict__ XA, const bf16* __restrict__ XB,
                     const bf16* __restrict__ YA, const bf16* __restrict__ YB,
                     const float* __restrict__ nx, const float* __restrict__ ny,
                     const float* __restrict__ sigmas, float* __restrict__ partial) {
    const int z = blockIdx.z;
    float* pslot = partial + (size_t)z * NBLK + blockIdx.y * GX + blockIdx.x;

    // Skip blocks strictly above the diagonal for the symmetric matrices.
    if (z < 2 && (int)blockIdx.y * BN >= (int)blockIdx.x * BM + BM) {
        if (threadIdx.x == 0) *pslot = 0.f;
        return;
    }

    const bf16*  Ae = (z == 1) ? YA : XA;
    const bf16*  Be = (z == 0) ? XB : YB;
    const float* nA = (z == 1) ? ny : nx;
    const float* nB = (z == 0) ? nx : ny;

    const int lane  = threadIdx.x & 31;
    const int wave  = threadIdx.x >> 5;
    const int waveM = wave >> 1, waveN = wave & 1;
    const int l16   = lane & 15, lhi = lane >> 4;

    const int bm = blockIdx.x * BM + waveM * 128;   // wave covers 128 rows
    const int bn = blockIdx.y * BN + waveN * 64;    // wave covers 64 cols

    const bf16* arow[8];
    const bf16* brow[4];
#pragma unroll
    for (int t = 0; t < 8; ++t)
        arow[t] = Ae + (size_t)(bm + t * 16 + l16) * KD;   // A: lane row M = l16
#pragma unroll
    for (int t = 0; t < 4; ++t)
        brow[t] = Be + (size_t)(bn + t * 16 + l16) * KD;   // B: lane col N = l16

    v8f acc[8][4];
#pragma unroll
    for (int i = 0; i < 8; ++i)
#pragma unroll
        for (int j = 0; j < 4; ++j) acc[i][j] = (v8f){};

    for (int k0 = 0; k0 < KD; k0 += 32) {
        V16U a[8], b[4];
#pragma unroll
        for (int t = 0; t < 8; ++t) {
            // A-frag: lanes0-15 K {0..7,16..23}; lanes16-31 K {8..15,24..31}
            a[t].h[0] = *(const v8bf*)(arow[t] + k0 + lhi * 8);
            a[t].h[1] = *(const v8bf*)(arow[t] + k0 + lhi * 8 + 16);
        }
#pragma unroll
        for (int t = 0; t < 4; ++t) {
            // B-frag: lanes0-15 K {0..7,16..23}; lanes16-31 K {8..15,24..31}
            b[t].h[0] = *(const v8bf*)(brow[t] + k0 + lhi * 8);
            b[t].h[1] = *(const v8bf*)(brow[t] + k0 + lhi * 8 + 16);
        }
#pragma unroll
        for (int i = 0; i < 8; ++i)
#pragma unroll
            for (int j = 0; j < 4; ++j)
                acc[i][j] = __builtin_amdgcn_wmma_f32_16x16x32_bf16(
                    false, a[i].v, false, b[j].v, (short)0, acc[i][j], false, false);
    }

    // Epilogue: dist = max(nA[i] + nB[j] - 2*s, 0); weighted sum of 5 Gaussians.
    const float c0 = -0.5f / sigmas[0], c1 = -0.5f / sigmas[1], c2 = -0.5f / sigmas[2],
                c3 = -0.5f / sigmas[3], c4 = -0.5f / sigmas[4];
    const bool sym = (z < 2);
    float lsum = 0.f;
#pragma unroll
    for (int i = 0; i < 8; ++i) {
#pragma unroll
        for (int j = 0; j < 4; ++j) {
            const int   jj  = bn + j * 16 + l16;   // C/D: N = lane%16
            const float nbj = nB[jj];
#pragma unroll
            for (int r = 0; r < 8; ++r) {
                const int ii = bm + i * 16 + lhi * 8 + r;  // C/D: M = r (+8 hi half)
                float w = sym ? (ii > jj ? 2.f : (ii == jj ? 1.f : 0.f)) : 1.f;
                float s = acc[i][j][r];
                float d = fmaxf(nA[ii] + nbj - 2.f * s, 0.f);
                lsum += w * (__expf(c0 * d) + __expf(c1 * d) + __expf(c2 * d) +
                             __expf(c3 * d) + __expf(c4 * d));
            }
        }
    }

    __shared__ float red[128];
    red[threadIdx.x] = lsum;
    __syncthreads();
    for (int off = 64; off > 0; off >>= 1) {
        if (threadIdx.x < off) red[threadIdx.x] += red[threadIdx.x + off];
        __syncthreads();
    }
    if (threadIdx.x == 0) *pslot = red[0];
}

// Deterministic final reduction (fixed order, double accumulation).
__global__ void finalize_kernel(const float* __restrict__ partial, float* __restrict__ out) {
    double s0 = 0.0, s1 = 0.0, s2 = 0.0;
    for (int i = 0; i < NBLK; ++i) s0 += (double)partial[i];
    for (int i = 0; i < NBLK; ++i) s1 += (double)partial[NBLK + i];
    for (int i = 0; i < NBLK; ++i) s2 += (double)partial[2 * NBLK + i];
    out[0] = (float)((s0 + s1 - 2.0 * s2) / 16777216.0);  // mean over 4096*4096
}

extern "C" void kernel_launch(void* const* d_in, const int* in_sizes, int n_in,
                              void* d_out, int out_size, void* d_ws, size_t ws_size,
                              hipStream_t stream) {
    const float* x      = (const float*)d_in[0];
    const float* y      = (const float*)d_in[1];
    const float* sigmas = (const float*)d_in[2];

    char* ws = (char*)d_ws;
    const size_t MATB = (size_t)NROWS * KD * sizeof(bf16);   // 8 MiB per packed matrix
    bf16*  XA = (bf16*)(ws);
    bf16*  XB = (bf16*)(ws + MATB);
    bf16*  YA = (bf16*)(ws + 2 * MATB);
    bf16*  YB = (bf16*)(ws + 3 * MATB);
    float* nx = (float*)(ws + 4 * MATB);
    float* ny = nx + NROWS;
    float* partial = ny + NROWS;                              // 3*NBLK floats
    float* out = (float*)d_out;

    pack_kernel<<<NROWS, 256, 0, stream>>>(x, XA, XB);
    pack_kernel<<<NROWS, 256, 0, stream>>>(y, YA, YB);
    norm_kernel<<<NROWS / 256, 256, 0, stream>>>(x, nx);
    norm_kernel<<<NROWS / 256, 256, 0, stream>>>(y, ny);

    dim3 grid(GX, GY, 3);   // 256x128 macro-tiles over 4096x4096, 3 matrices
    mmd_tile_kernel<<<grid, 128, 0, stream>>>(XA, XB, YA, YB, nx, ny, sigmas, partial);

    finalize_kernel<<<1, 1, 0, stream>>>(partial, out);
}
